// InstanSeg_Torchscript_54125177864462
// MI455X (gfx1250) — matrix operand order repaired and checked by reading, not compiled, not run on hardware
//
#include <hip/hip_runtime.h>
#include <hip/hip_bf16.h>
#include <math.h>

// ---------------------------------------------------------------------------
// InstanSeg label-map + connected-components remap for MI455X (gfx1250).
//
// Phase 1 (memory bound, ~100MB traffic): per-key winner via a single
//   atomicMax on packed u64 = (prob_bits << 32) | ~pixel_index.
//   prob>=0.53 => prob_bits>0 => packed!=0, so 0 is the "empty" sentinel.
// Phase 2 (WMMA): P = (A+I)^10 via repeated squaring using
//   v_wmma_f32_16x16x32_bf16 (bf16 inputs / f32 accumulate). The >1
//   threshold is exact under bf16 rounding (walk counts <=2 are built only
//   from factors <=2, all exact in bf16; counts >=3 stay > 1).
// Phase 3: replicate reference argmax/remap quirks, then gather to output.
// ---------------------------------------------------------------------------

typedef __attribute__((ext_vector_type(16))) __bf16 v16bf;
typedef __attribute__((ext_vector_type(8)))  float  v8f;

#define MASK_THR 0.53f

__global__ void zero_map_kernel(unsigned long long* __restrict__ map, int n) {
  int i = blockIdx.x * blockDim.x + threadIdx.x;
  if (i < n) map[i] = 0ull;
}

__global__ void scatter_kernel(const float* __restrict__ prob,
                               const int* __restrict__ coords,   // [2, npix]
                               const int* __restrict__ size1_ptr,
                               unsigned long long* __restrict__ map,
                               int npix, int map_elems) {
  int i = blockIdx.x * blockDim.x + threadIdx.x;
  if (i >= npix) return;
  float p = prob[i];
  if (!(p >= MASK_THR)) return;                      // sub-threshold -> dropped
  long long key = (long long)coords[i] * (long long)size1_ptr[0]
                + (long long)coords[npix + i];
  if (key < 0 || key >= (long long)map_elems) return;
  // max prob wins; equal prob -> smaller original index wins (~i is larger)
  unsigned long long packed =
      ((unsigned long long)__float_as_uint(p) << 32) |
      (unsigned long long)(~(unsigned int)i);
  atomicMax(&map[key], packed);
}

__global__ void build_m_kernel(const float* __restrict__ adj,
                               float* __restrict__ M, int n) {
  int i = blockIdx.x * blockDim.x + threadIdx.x;
  if (i >= n * n) return;
  int r = i / n, c = i - r * n;
  M[i] = adj[i] + (r == c ? 1.0f : 0.0f);
}

// One wave32 computes one 16x16 tile of C = A*B (row-major n x n, n % 32 == 0).
// Fragment layouts per CDNA5 ISA 7.12.2 (16-bit A 16x32 / B 32x16 / f32 C 16x16).
__global__ __launch_bounds__(32) void wmma_mm_kernel(const float* __restrict__ A,
                                                     const float* __restrict__ B,
                                                     float* __restrict__ C, int n) {
  const int lane = threadIdx.x;         // 0..31
  const int m    = lane & 15;           // row (A) / col (B,D) within tile
  const int half = (lane >> 4) & 1;     // lane group 0..15 vs 16..31
  const int mt   = blockIdx.y << 4;
  const int nt   = blockIdx.x << 4;

  v8f acc = {};
  for (int k0 = 0; k0 < n; k0 += 32) {
    v16bf a, b;
#pragma unroll
    for (int v = 0; v < 8; ++v) {
      // A 16x32 bf16: lanes0-15 v0-3 K=0..7, v4-7 K=16..23; lanes16-31 +8
      const int ka = ((v & 4) ? 16 : 0) + half * 8 + (v & 3) * 2;
      const float2 av = *(const float2*)&A[(size_t)(mt + m) * n + (k0 + ka)];
      a[2 * v]     = (__bf16)av.x;
      a[2 * v + 1] = (__bf16)av.y;
      // B 32x16 bf16: lanes0-15 v0-3 K=0..7, v4-7 K=16..23; lanes16-31 +8
      const int kb = ((v & 4) ? 16 : 0) + half * 8 + (v & 3) * 2;
      b[2 * v]     = (__bf16)B[(size_t)(k0 + kb)     * n + (nt + m)];
      b[2 * v + 1] = (__bf16)B[(size_t)(k0 + kb + 1) * n + (nt + m)];
    }
    // (neg_a, A, neg_b, B, c_mod, C, reuse_a, reuse_b)
    acc = __builtin_amdgcn_wmma_f32_16x16x32_bf16(
        false, a, false, b, (short)0, acc, false, false);
  }
#pragma unroll
  for (int r = 0; r < 8; ++r)           // C/D: vgpr r -> M = r + half*8, lane -> N
    C[(size_t)(mt + r + half * 8) * n + (nt + m)] = acc[r];
}

// Replicates:
//   out_col_idx = ((P>1) - eye) * rowidx
//   maxes = argmax(out_col_idx,0) * (max(out_col_idx,0) > 0)
//   maxes = maximum(maxes+1, arange+1);  remap = [0, maxes...]
__global__ void cc_kernel(const float* __restrict__ P,
                          float* __restrict__ remap, int n) {
  int j = blockIdx.x * blockDim.x + threadIdx.x;
  if (j >= n) return;
  float best = 0.0f; int arg = 0;
  for (int i = 0; i < n; ++i) {
    const bool gt = P[(size_t)i * n + j] > 1.0f;
    float v;
    if (i == j) v = gt ? 0.0f : -(float)j;   // (1-1)*j or (0-1)*j
    else        v = gt ? (float)i : 0.0f;
    if (i == 0)           { best = v; arg = 0; }
    else if (v > best)    { best = v; arg = i; }   // first-occurrence argmax
  }
  const int maxes = (best > 0.0f) ? arg : 0;
  remap[j + 1] = fmaxf((float)(maxes + 1), (float)(j + 1));
  if (j == 0) remap[0] = 0.0f;
}

__global__ void final_kernel(const unsigned long long* __restrict__ map,
                             const float* __restrict__ remap,
                             float* __restrict__ out, int map_elems, int hw) {
  int i = blockIdx.x * blockDim.x + threadIdx.x;
  if (i >= map_elems) return;
  const unsigned long long w = map[i];
  int label = 0;
  if (w != 0ull) {
    const unsigned int idx = ~(unsigned int)(w & 0xFFFFFFFFull);
    label = (int)(idx / (unsigned int)hw) + 1;     // instance id from flat index
  }
  out[i] = remap[label];
}

extern "C" void kernel_launch(void* const* d_in, const int* in_sizes, int n_in,
                              void* d_out, int out_size, void* d_ws, size_t ws_size,
                              hipStream_t stream) {
  const float* prob   = (const float*)d_in[0];
  const int*   coords = (const int*)d_in[1];
  const float* adj    = (const float*)d_in[2];
  const int*   size1p = (const int*)d_in[4];
  float*       out    = (float*)d_out;

  const int npix = in_sizes[0];                              // N*1*h*w
  const int n    = (int)(sqrt((double)in_sizes[2]) + 0.5);   // adjacency is N x N
  const int hw   = npix / n;                                 // h*w per instance
  const int map_elems = out_size;                            // size0*size1

  // Workspace layout
  unsigned long long* map  = (unsigned long long*)d_ws;
  float* bufA  = (float*)((char*)d_ws + (size_t)map_elems * sizeof(unsigned long long));
  float* bufB  = bufA + (size_t)n * n;
  float* bufC  = bufB + (size_t)n * n;
  float* remap = bufC + (size_t)n * n;                       // n+1 floats

  const int T = 256;

  // Phase 1: winner scatter
  zero_map_kernel<<<(map_elems + T - 1) / T, T, 0, stream>>>(map, map_elems);
  scatter_kernel<<<(npix + T - 1) / T, T, 0, stream>>>(prob, coords, size1p,
                                                       map, npix, map_elems);

  // Phase 2: P = (A+I)^10 via repeated squaring (M^8 @ M^2), WMMA bf16
  build_m_kernel<<<(n * n + T - 1) / T, T, 0, stream>>>(adj, bufA, n);
  dim3 g(n / 16, n / 16);
  wmma_mm_kernel<<<g, 32, 0, stream>>>(bufA, bufA, bufB, n);   // B = M^2
  wmma_mm_kernel<<<g, 32, 0, stream>>>(bufB, bufB, bufC, n);   // C = M^4
  wmma_mm_kernel<<<g, 32, 0, stream>>>(bufC, bufC, bufA, n);   // A = M^8
  wmma_mm_kernel<<<g, 32, 0, stream>>>(bufA, bufB, bufC, n);   // C = M^10

  // Phase 3: component remap + gather
  cc_kernel<<<(n + T - 1) / T, T, 0, stream>>>(bufC, remap, n);
  final_kernel<<<(map_elems + T - 1) / T, T, 0, stream>>>(map, remap, out,
                                                          map_elems, hw);
}
